// SwitchExperts_88742614270196
// MI455X (gfx1250) — compile-verified, matrix operand order checked by reading, not run on hardware
//
#include <hip/hip_runtime.h>
#include <hip/hip_bf16.h>

// ---------------------------------------------------------------------------
// Switch-MoE (top-1) for MI455X / gfx1250, wave32 + WMMA bf16.
//   T=8192 tokens, D=1024, E=8 experts.
// Pipeline: router -> per-expert token grouping -> 3 grouped bf16 WMMA GEMMs.
// GEMM: 128x128x32 tile, 8 waves, double-buffered LDS with register staging;
// each wave = 32x64 macro-tile = 2x4 v_wmma_f32_16x16x32_bf16 per K-step.
// bf16 packing via v_perm_b32 (round-half-up), A-row guards replaced by clamp.
// Scratch (d_ws, ~112.1 MB): counters/offsets/ids/perm + xg(16MB) h1(64MB) h2(32MB)
// ---------------------------------------------------------------------------

#define TOK    8192
#define DMODEL 1024
#define NEXP   8
#define BM     128
#define BN     128
#define BK     32
#define MTMAX  (TOK / BM)   // 64 m-tiles covers the worst case (all tokens -> one expert)

typedef __attribute__((ext_vector_type(16))) __bf16 v16bf;
typedef __attribute__((ext_vector_type(8)))  float  v8f;

union FragBF { unsigned int u[8]; v16bf v; };

static __device__ __forceinline__ unsigned int f32_bits(float f) {
  union { float f; unsigned int u; } x; x.f = f; return x.u;
}
// round-half-up bf16 (<= 1/2 ulp from RNE; inputs finite)
static __device__ __forceinline__ unsigned short f32_to_bf16(float f) {
  return (unsigned short)((f32_bits(f) + 0x8000u) >> 16);
}
// pack two floats -> bf16x2 with a single v_perm_b32 splicing the high halves
static __device__ __forceinline__ unsigned int pack_bf16x2(float lo, float hi) {
  const unsigned int lu = f32_bits(lo) + 0x8000u;
  const unsigned int hu = f32_bits(hi) + 0x8000u;
  // bytes: S0 (hu) = idx 4..7, S1 (lu) = idx 0..3 ; want {hu3,hu2,lu3,lu2}
  return __builtin_amdgcn_perm(hu, lu, 0x07060302u);
}
static __device__ __forceinline__ float gelu_exact(float v) {
  return 0.5f * v * (1.0f + erff(v * 0.70710678118654752f));
}

// ---------------------------------------------------------------------------
__global__ void moe_init_kernel(int* counts) {
  if (threadIdx.x < NEXP) counts[threadIdx.x] = 0;
}

// ---------------------------------------------------------------------------
// router: logits = x @ Wr + br ; expert_id = argmax (first max, like jnp.argmax)
// one wave32 per token, Wr staged in LDS
// ---------------------------------------------------------------------------
__global__ __launch_bounds__(256)
void moe_router_kernel(const float* __restrict__ x, const float* __restrict__ Wr,
                       const float* __restrict__ br, int* __restrict__ expert_id,
                       int* __restrict__ counts) {
  __shared__ float sWr[DMODEL * NEXP];
  for (int i = threadIdx.x; i < DMODEL * NEXP; i += 256) sWr[i] = Wr[i];
  __syncthreads();

  const int lane = threadIdx.x & 31;
  const int wave = threadIdx.x >> 5;
  const int tok  = blockIdx.x * 8 + wave;
  if (tok >= TOK) return;

  float acc[NEXP];
#pragma unroll
  for (int e = 0; e < NEXP; ++e) acc[e] = 0.0f;

  const float* xr = x + (size_t)tok * DMODEL;
  for (int d = lane; d < DMODEL; d += 32) {
    const float xv = xr[d];
#pragma unroll
    for (int e = 0; e < NEXP; ++e) acc[e] += xv * sWr[d * NEXP + e];
  }
#pragma unroll
  for (int e = 0; e < NEXP; ++e) {
#pragma unroll
    for (int off = 16; off > 0; off >>= 1) acc[e] += __shfl_down(acc[e], off);
  }
  if (lane == 0) {
    int   best = 0;
    float bv   = acc[0] + br[0];
#pragma unroll
    for (int e = 1; e < NEXP; ++e) {
      const float v = acc[e] + br[e];
      if (v > bv) { bv = v; best = e; }
    }
    expert_id[tok] = best;
    atomicAdd(&counts[best], 1);
  }
}

// ---------------------------------------------------------------------------
__global__ void moe_scan_kernel(const int* __restrict__ counts,
                                int* __restrict__ offsets, int* __restrict__ cursors) {
  if (threadIdx.x == 0) {
    int run = 0;
    for (int e = 0; e < NEXP; ++e) { offsets[e] = run; cursors[e] = run; run += counts[e]; }
  }
}

__global__ __launch_bounds__(256)
void moe_scatter_kernel(const int* __restrict__ expert_id, int* __restrict__ cursors,
                        int* __restrict__ perm) {
  const int t = blockIdx.x * 256 + threadIdx.x;
  if (t >= TOK) return;
  const int e   = expert_id[t];
  const int pos = atomicAdd(&cursors[e], 1);
  perm[pos] = t;
}

__global__ __launch_bounds__(256)
void moe_gather_kernel(const float* __restrict__ x, const int* __restrict__ perm,
                       unsigned short* __restrict__ xg) {
  const int gid = blockIdx.x * 256 + threadIdx.x;   // T*256 threads, 4 elems each
  const int row = gid >> 8;
  const int c   = (gid & 255) << 2;
  if (row >= TOK) return;
  const int tok = perm[row];
  const float4 v = *reinterpret_cast<const float4*>(&x[(size_t)tok * DMODEL + c]);
  uint2 p;
  p.x = pack_bf16x2(v.x, v.y);
  p.y = pack_bf16x2(v.z, v.w);
  *reinterpret_cast<uint2*>(&xg[(size_t)row * DMODEL + c]) = p;
}

// ---------------------------------------------------------------------------
// grouped GEMM, double-buffered.  MODE 0: GELU -> bf16 to Hout.
// MODE 1: +bias -> fp32 scatter to Yout rows via perm.
// A fragments per ISA 7.12.2 (16-bit A 16x32); B staged [n][k-pairs] in LDS so
// B fragments use the mirrored (N<->M) layout; packed uint = (k even, k odd).
// Out-of-segment rows are clamped on load (results masked on store), so the
// inner loop has no divergent guards.
// ---------------------------------------------------------------------------
template <int KDIM, int NDIM, int MODE>
__global__ __launch_bounds__(256)
void moe_gemm_kernel(const unsigned short* __restrict__ Abf,   // [TOK, KDIM] bf16, permuted rows
                     const float* __restrict__ W,              // [E, KDIM, NDIM] fp32
                     const float* __restrict__ bias,           // [E, NDIM] fp32
                     const int* __restrict__ offsets,
                     const int* __restrict__ counts,
                     const int* __restrict__ perm,
                     unsigned short* __restrict__ Hout,        // bf16 (MODE 0)
                     float* __restrict__ Yout)                 // fp32 (MODE 1)
{
  const int e  = blockIdx.x / MTMAX;
  const int mt = blockIdx.x % MTMAX;
  const int cnt = counts[e];
  if (mt * BM >= cnt) return;

  const int segStart = offsets[e];
  const int rowBase  = segStart + mt * BM;
  const int ncolBase = blockIdx.y * BN;

  __shared__ unsigned int As[2][BM * 18];          // 128 x (32+4 pad) bf16, ping-pong
  __shared__ unsigned int Bs[2][BN * 18];          // [n][k/2] packed pairs, ping-pong

  const int tid   = threadIdx.x;
  const int lane  = tid & 31;
  const int l16   = lane & 15;
  const int lhalf = lane >> 4;
  const int wave  = tid >> 5;
  const int wm    = (wave & 3) * 32;               // 4 waves along M
  const int wn    = (wave >> 2) * 64;              // 2 waves along N

  const float* We = W + (size_t)e * KDIM * NDIM;

  // Per-thread staging coordinates (fixed across K-steps).  Rows clamped: any
  // clamped row's products are discarded by the store mask.
  const int arow0 = (tid + 0)   >> 2, acc0 = ((tid + 0)   & 3) * 8;
  const int arow1 = (tid + 256) >> 2, acc1 = ((tid + 256) & 3) * 8;
  const int bkp0  = (tid + 0)   >> 5, bc40 = ((tid + 0)   & 31) * 4;
  const int bkp1  = (tid + 256) >> 5, bc41 = ((tid + 256) & 31) * 4;

  const int rowC0 = min(rowBase + arow0, TOK - 1);
  const int rowC1 = min(rowBase + arow1, TOK - 1);
  const unsigned short* aP0 = Abf + (size_t)rowC0 * KDIM + acc0;
  const unsigned short* aP1 = Abf + (size_t)rowC1 * KDIM + acc1;
  const float* bP0 = We + (size_t)(2 * bkp0) * NDIM + ncolBase + bc40;
  const float* bP1 = We + (size_t)(2 * bkp1) * NDIM + ncolBase + bc41;

  uint4  aR[2];
  float4 bR[4];

  auto stage_load = [&](int k0) {
    aR[0] = *reinterpret_cast<const uint4*>(aP0 + k0);
    aR[1] = *reinterpret_cast<const uint4*>(aP1 + k0);
    bR[0] = *reinterpret_cast<const float4*>(bP0 + (size_t)k0 * NDIM);
    bR[1] = *reinterpret_cast<const float4*>(bP0 + (size_t)(k0 + 1) * NDIM);
    bR[2] = *reinterpret_cast<const float4*>(bP1 + (size_t)k0 * NDIM);
    bR[3] = *reinterpret_cast<const float4*>(bP1 + (size_t)(k0 + 1) * NDIM);
  };
  auto stage_store = [&](int buf) {
    unsigned int* a0 = &As[buf][arow0 * 18 + (acc0 >> 1)];
    a0[0] = aR[0].x; a0[1] = aR[0].y; a0[2] = aR[0].z; a0[3] = aR[0].w;
    unsigned int* a1 = &As[buf][arow1 * 18 + (acc1 >> 1)];
    a1[0] = aR[1].x; a1[1] = aR[1].y; a1[2] = aR[1].z; a1[3] = aR[1].w;
    Bs[buf][(bc40 + 0) * 18 + bkp0] = pack_bf16x2(bR[0].x, bR[1].x);
    Bs[buf][(bc40 + 1) * 18 + bkp0] = pack_bf16x2(bR[0].y, bR[1].y);
    Bs[buf][(bc40 + 2) * 18 + bkp0] = pack_bf16x2(bR[0].z, bR[1].z);
    Bs[buf][(bc40 + 3) * 18 + bkp0] = pack_bf16x2(bR[0].w, bR[1].w);
    Bs[buf][(bc41 + 0) * 18 + bkp1] = pack_bf16x2(bR[2].x, bR[3].x);
    Bs[buf][(bc41 + 1) * 18 + bkp1] = pack_bf16x2(bR[2].y, bR[3].y);
    Bs[buf][(bc41 + 2) * 18 + bkp1] = pack_bf16x2(bR[2].z, bR[3].z);
    Bs[buf][(bc41 + 3) * 18 + bkp1] = pack_bf16x2(bR[2].w, bR[3].w);
  };

  v8f acc[2][4];
#pragma unroll
  for (int mi = 0; mi < 2; ++mi)
#pragma unroll
    for (int ni = 0; ni < 4; ++ni) {
      v8f z = {0.f, 0.f, 0.f, 0.f, 0.f, 0.f, 0.f, 0.f};
      acc[mi][ni] = z;
    }

  constexpr int KSTEPS = KDIM / BK;
  stage_load(0);
  stage_store(0);
  __syncthreads();

  int cur = 0;
  for (int ks = 0; ks < KSTEPS; ++ks) {
    const bool more = (ks + 1 < KSTEPS);
    if (more) stage_load((ks + 1) * BK);           // next tile's global loads in flight
    if (ks + 2 < KSTEPS)                           // hint k+2 weights toward GL2
      __builtin_prefetch(&We[(size_t)((ks + 2) * BK + (tid >> 3)) * NDIM + ncolBase + (tid & 7) * 16], 0, 1);

    // ---- fragments from current buffer + 8 WMMAs -------------------------
    const unsigned int* Ac = As[cur];
    const unsigned int* Bc = Bs[cur];
    FragBF a[2], b[4];
#pragma unroll
    for (int mi = 0; mi < 2; ++mi) {
      const int m = wm + mi * 16 + l16;
#pragma unroll
      for (int v = 0; v < 8; ++v) {
        const int kph = ((v & 4) ? 8 : 0) + (lhalf ? 4 : 0) + (v & 3);  // uint index of k-pair
        a[mi].u[v] = Ac[m * 18 + kph];
      }
    }
#pragma unroll
    for (int ni = 0; ni < 4; ++ni) {
      const int n = wn + ni * 16 + l16;
#pragma unroll
      for (int v = 0; v < 8; ++v) {
        const int kph = ((v & 4) ? 8 : 0) + (lhalf ? 4 : 0) + (v & 3);
        b[ni].u[v] = Bc[n * 18 + kph];
      }
    }
#pragma unroll
    for (int mi = 0; mi < 2; ++mi)
#pragma unroll
      for (int ni = 0; ni < 4; ++ni)
        acc[mi][ni] = __builtin_amdgcn_wmma_f32_16x16x32_bf16(
            false, a[mi].v, false, b[ni].v, (short)0, acc[mi][ni], false, false);

    if (more) stage_store(cur ^ 1);                // fill alternate buffer
    __syncthreads();
    cur ^= 1;
  }

  // ---- epilogue: bias (+GELU) and store, masked to the expert segment ----
  const float* biasE = bias + (size_t)e * NDIM;
  float bcol[4];
#pragma unroll
  for (int ni = 0; ni < 4; ++ni) bcol[ni] = biasE[ncolBase + wn + ni * 16 + l16];

#pragma unroll
  for (int mi = 0; mi < 2; ++mi) {
#pragma unroll
    for (int r = 0; r < 8; ++r) {
      const int rowL   = wm + mi * 16 + (lhalf << 3) + r;    // C layout: lanes>=16 -> M+8
      const int rowSeg = mt * BM + rowL;
      if (rowSeg < cnt) {
        if (MODE == 0) {
          unsigned short* hrow = &Hout[(size_t)(segStart + rowSeg) * NDIM];
#pragma unroll
          for (int ni = 0; ni < 4; ++ni) {
            const int col = ncolBase + wn + ni * 16 + l16;
            hrow[col] = f32_to_bf16(gelu_exact(acc[mi][ni][r] + bcol[ni]));
          }
        } else {
          const int tok = perm[segStart + rowSeg];
          float* yrow = &Yout[(size_t)tok * NDIM];
#pragma unroll
          for (int ni = 0; ni < 4; ++ni) {
            const int col = ncolBase + wn + ni * 16 + l16;
            yrow[col] = acc[mi][ni][r] + bcol[ni];
          }
        }
      }
    }
  }
}

// ---------------------------------------------------------------------------
extern "C" void kernel_launch(void* const* d_in, const int* in_sizes, int n_in,
                              void* d_out, int out_size, void* d_ws, size_t ws_size,
                              hipStream_t stream) {
  (void)in_sizes; (void)n_in; (void)out_size; (void)ws_size;

  const float* x  = (const float*)d_in[0];
  const float* Wr = (const float*)d_in[1];
  const float* br = (const float*)d_in[2];
  const float* W1 = (const float*)d_in[3];
  const float* b1 = (const float*)d_in[4];
  const float* W2 = (const float*)d_in[5];
  const float* b2 = (const float*)d_in[6];
  const float* W3 = (const float*)d_in[7];
  const float* b3 = (const float*)d_in[8];
  float* out = (float*)d_out;

  char* ws = (char*)d_ws;
  int* counts    = (int*)(ws + 0);
  int* offsets   = (int*)(ws + 256);
  int* cursors   = (int*)(ws + 512);
  int* expert_id = (int*)(ws + 1024);
  int* perm      = (int*)(ws + 1024 + 4 * TOK);
  unsigned short* xg = (unsigned short*)(ws + 1024 + 8 * TOK);
  unsigned short* h1 = xg + (size_t)TOK * DMODEL;
  unsigned short* h2 = h1 + (size_t)TOK * (4 * DMODEL);

  moe_init_kernel<<<1, 32, 0, stream>>>(counts);
  moe_router_kernel<<<TOK / 8, 256, 0, stream>>>(x, Wr, br, expert_id, counts);
  moe_scan_kernel<<<1, 32, 0, stream>>>(counts, offsets, cursors);
  moe_scatter_kernel<<<TOK / 256, 256, 0, stream>>>(expert_id, cursors, perm);
  moe_gather_kernel<<<TOK, 256, 0, stream>>>(x, perm, xg);

  dim3 blk(256);
  dim3 g1(NEXP * MTMAX, (4 * DMODEL) / BN);
  moe_gemm_kernel<DMODEL, 4 * DMODEL, 0><<<g1, blk, 0, stream>>>(
      xg, W1, b1, offsets, counts, perm, h1, nullptr);
  dim3 g2(NEXP * MTMAX, (2 * DMODEL) / BN);
  moe_gemm_kernel<4 * DMODEL, 2 * DMODEL, 0><<<g2, blk, 0, stream>>>(
      h1, W2, b2, offsets, counts, perm, h2, nullptr);
  dim3 g3(NEXP * MTMAX, DMODEL / BN);
  moe_gemm_kernel<2 * DMODEL, DMODEL, 1><<<g3, blk, 0, stream>>>(
      h2, W3, b3, offsets, counts, perm, nullptr, out);
}
